// SequenceDecoder_88673894793898
// MI455X (gfx1250) — matrix operand order repaired and checked by reading, not compiled, not run on hardware
//
#include <hip/hip_runtime.h>

// ---------------- problem constants (match reference) ----------------
#define BB   64
#define LAT  512
#define HH   1024
#define G4H  4096
#define VV   8192
#define TT   128
#define NWG  64          // persistent workgroups (must all be co-resident)
#define NTHR 256         // 8 waves of 32
#define NWAVES (NWG * (NTHR / 32))   // 512

typedef __bf16 bf16;
typedef __attribute__((ext_vector_type(16))) __bf16 v16bf;
typedef __attribute__((ext_vector_type(8)))  float  v8f;
typedef __attribute__((ext_vector_type(4)))  unsigned u32x4;
typedef __attribute__((ext_vector_type(2)))  unsigned u32x2;
typedef __attribute__((ext_vector_type(4)))  int      v4i;

union FragU { v16bf v; u32x4 q[2]; };

__device__ __forceinline__ unsigned short f2bf(float f) {
    unsigned u = __float_as_uint(f);
    u += 0x7FFFu + ((u >> 16) & 1u);          // round-to-nearest-even
    return (unsigned short)(u >> 16);
}
__device__ __forceinline__ float sigmoidf(float x) { return 1.0f / (1.0f + __expf(-x)); }

// Accumulate NT adjacent 16x16 f32 tiles:  acc[j] += A(16xK) * W[n0+16j : +16, 0:K]^T
// a_base: row-0 of the A tile (LDS or global), bf16 row-major with leading dim a_ld.
// W: row-major [N, w_ld] bf16 (K contiguous -> two b128 per fragment per lane).
template<int NT>
__device__ __forceinline__ void gemm_tiles(v8f* acc,
                                           const unsigned short* __restrict__ a_base, int a_ld,
                                           const unsigned short* __restrict__ W,      int w_ld,
                                           int K, int n0, int lane) {
    const int mn = lane & 15;
    // A-fragment lane layout (ISA 16-bit A): lanes0-15 K {0..7,16..23}, lanes16-31 K {8..15,24..31}
    const unsigned short* ap = a_base + mn * a_ld + ((lane & 16) ? 8 : 0);
    // B-fragment lane layout (same as A): lanes0-15 K {0..7,16..23}, lanes16-31 K {8..15,24..31}
    const unsigned short* bp0 = W + (size_t)(n0 + mn) * w_ld + ((lane & 16) ? 8 : 0);
    for (int k0 = 0; k0 < K; k0 += 32) {
        FragU a;
        a.q[0] = *(const u32x4*)(ap);
        a.q[1] = *(const u32x4*)(ap + 16);
        ap += 32;
        #pragma unroll
        for (int j = 0; j < NT; ++j) {
            FragU b;
            const unsigned short* bp = bp0 + (size_t)(16 * j) * w_ld;
            b.q[0] = *(const u32x4*)(bp);
            b.q[1] = *(const u32x4*)(bp + 16);
            acc[j] = __builtin_amdgcn_wmma_f32_16x16x32_bf16(false, a.v, false, b.v,
                                                             (short)0, acc[j], false, false);
        }
        bp0 += 32;
        __builtin_prefetch(bp0, 0, 1);         // global_prefetch_b8 on the weight stream
    }
}

// Copy a contiguous 16-row bf16 activation slice into LDS.
// Preferred path: CDNA5 async Global->LDS DMA (no VGPR round-trip, tracked by ASYNCcnt).
__device__ __forceinline__ void stage_slice(unsigned short* lds, const unsigned short* src,
                                            int elems, int tid) {
#if __has_builtin(__builtin_amdgcn_global_load_async_to_lds_b128) && \
    __has_builtin(__builtin_amdgcn_s_wait_asynccnt)
    typedef __attribute__((address_space(1))) v4i g_v4i;   // global int4
    typedef __attribute__((address_space(3))) v4i l_v4i;   // LDS int4
    for (int i = tid; i < elems / 8; i += NTHR) {   // 16 bytes per lane per issue
        __builtin_amdgcn_global_load_async_to_lds_b128(
            (g_v4i*)(src + i * 8), (l_v4i*)(lds + i * 8), 0, 0);
    }
    __builtin_amdgcn_s_wait_asynccnt(0);
#else
    const u32x4* s = (const u32x4*)src;
    u32x4* d = (u32x4*)lds;
    for (int i = tid; i < elems / 8; i += NTHR) d[i] = s[i];
#endif
}

// Monotone-counter grid barrier (all NWG blocks co-resident).
__device__ __forceinline__ void grid_barrier(unsigned* cnt, unsigned& phase) {
    __syncthreads();
    if (threadIdx.x == 0) {
        __builtin_amdgcn_fence(__ATOMIC_RELEASE, "agent");
        __hip_atomic_fetch_add(cnt, 1u, __ATOMIC_RELAXED, __HIP_MEMORY_SCOPE_AGENT);
        const unsigned target = (phase + 1u) * (unsigned)NWG;
        while (__hip_atomic_load(cnt, __ATOMIC_RELAXED, __HIP_MEMORY_SCOPE_AGENT) < target)
            __builtin_amdgcn_s_sleep(2);
    }
    __syncthreads();
    __builtin_amdgcn_fence(__ATOMIC_ACQUIRE, "agent");
    phase++;
}

__global__ void f32_to_bf16_kernel(const float* __restrict__ src,
                                   unsigned short* __restrict__ dst, int n) {
    for (int i = blockIdx.x * blockDim.x + threadIdx.x; i < n; i += gridDim.x * blockDim.x)
        dst[i] = f2bf(src[i]);
}

__global__ void __launch_bounds__(NTHR)
lstm_decoder_kernel(const unsigned short* __restrict__ zb,
                    const unsigned short* __restrict__ Eb,
                    const unsigned short* __restrict__ Wihb,
                    const unsigned short* __restrict__ Whhb,
                    const unsigned short* __restrict__ Wlatb,
                    const unsigned short* __restrict__ Woutb,
                    const float* __restrict__ b_ih, const float* __restrict__ b_hh,
                    const float* __restrict__ b_lat, const float* __restrict__ b_out,
                    float* __restrict__ gates, float* __restrict__ cbuf,
                    unsigned short* __restrict__ hb, unsigned short* __restrict__ xb,
                    float* __restrict__ out, unsigned* __restrict__ bar, int T) {
    __shared__ unsigned short lds_act[16 * HH];   // 32 KB activation slice (x or h)
    __shared__ float smax[NTHR];
    __shared__ int   sidx[NTHR];
    __shared__ int   s_tok;

    const int tid  = threadIdx.x;
    const int lane = tid & 31;
    const int wave = blockIdx.x * (NTHR / 32) + (tid >> 5);   // 0..511
    unsigned phase = 0;

    // ---------------- h0 = z @ W_lat^T + b_lat ; c0 = 0 ----------------
    // 256 tiles as 128 groups of NT=2; waves 0..127 active; A straight from global (z is tiny).
    if (wave < (BB / 16) * (HH / 16) / 2) {
        const int m0 = (wave >> 5) * 16;
        const int n0 = (wave & 31) * 32;
        v8f acc[2] = {};
        gemm_tiles<2>(acc, zb + m0 * LAT, LAT, Wlatb, LAT, LAT, n0, lane);
        #pragma unroll
        for (int j = 0; j < 2; ++j) {
            const int n = n0 + 16 * j + (lane & 15);
            const float bias = b_lat[n];
            #pragma unroll
            for (int v = 0; v < 8; ++v) {
                const int m = m0 + v + ((lane & 16) ? 8 : 0);
                const float hv = acc[j][v] + bias;
                hb[m * HH + n]   = f2bf(hv);
                cbuf[m * HH + n] = 0.0f;
            }
        }
    }
    // ---------------- x0 = E[token 0] ----------------
    if (blockIdx.x < BB) {
        const int b = blockIdx.x;
        const int j = tid * 4;                       // 256 threads * 4 bf16 = 1024
        *(u32x2*)(xb + b * HH + j) = *(const u32x2*)(Eb + j);
    }
    grid_barrier(bar, phase);

    for (int t = 0; t < T; ++t) {
        // ---- gates = x@W_ih^T + h@W_hh^T + b_ih + b_hh   [64 x 4096] ----
        // 1024 tiles = 512 groups of NT=2 -> exactly one group per wave.
        // wave>>7 is uniform per WG -> the whole WG shares one m0: stage its A slice in LDS.
        {
            const int m0 = (wave >> 7) * 16;
            const int n0 = (wave & 127) * 32;
            stage_slice(lds_act, xb + m0 * HH, 16 * HH, tid);
            __syncthreads();
            v8f acc[2] = {};
            gemm_tiles<2>(acc, lds_act, HH, Wihb, HH, HH, n0, lane);
            __syncthreads();
            stage_slice(lds_act, hb + m0 * HH, 16 * HH, tid);
            __syncthreads();
            gemm_tiles<2>(acc, lds_act, HH, Whhb, HH, HH, n0, lane);
            #pragma unroll
            for (int j = 0; j < 2; ++j) {
                const int n = n0 + 16 * j + (lane & 15);
                const float bias = b_ih[n] + b_hh[n];
                #pragma unroll
                for (int v = 0; v < 8; ++v) {
                    const int m = m0 + v + ((lane & 16) ? 8 : 0);
                    gates[m * G4H + n] = acc[j][v] + bias;
                }
            }
        }
        grid_barrier(bar, phase);

        // ---- elementwise LSTM cell ----
        for (int e = blockIdx.x * NTHR + tid; e < BB * HH; e += NWG * NTHR) {
            const int b = e >> 10, j = e & (HH - 1);
            const float* g = gates + b * G4H;
            const float ig = sigmoidf(g[j]);
            const float fg = sigmoidf(g[HH + j]);
            const float gg = tanhf(g[2 * HH + j]);
            const float og = sigmoidf(g[3 * HH + j]);
            const float cv = fg * cbuf[e] + ig * gg;
            const float hv = og * tanhf(cv);
            cbuf[e] = cv;
            hb[e]   = f2bf(hv);
        }
        grid_barrier(bar, phase);

        // ---- logits = h @ W_out^T + b_out  -> out[b][t][v] ----
        // 2048 tiles = 512 groups of NT=4 -> one group per wave; same WG-uniform m0.
        {
            const int m0 = (wave >> 7) * 16;
            const int n0 = (wave & 127) * 64;
            stage_slice(lds_act, hb + m0 * HH, 16 * HH, tid);
            __syncthreads();
            v8f acc[4] = {};
            gemm_tiles<4>(acc, lds_act, HH, Woutb, HH, HH, n0, lane);
            __syncthreads();
            #pragma unroll
            for (int j = 0; j < 4; ++j) {
                const int n = n0 + 16 * j + (lane & 15);
                const float bias = b_out[n];
                #pragma unroll
                for (int v = 0; v < 8; ++v) {
                    const int m = m0 + v + ((lane & 16) ? 8 : 0);
                    out[((size_t)m * T + t) * VV + n] = acc[j][v] + bias;
                }
            }
        }
        grid_barrier(bar, phase);

        // ---- argmax over V per batch row (first-max) + x = E[tok] ----
        if (blockIdx.x < BB) {
            const int b = blockIdx.x;
            const float* row = out + ((size_t)b * T + t) * VV;
            float best = -3.402823466e38f; int bidx = 0;
            const int j0 = tid * (VV / NTHR);
            for (int j = 0; j < VV / NTHR; ++j) {
                const float vv = row[j0 + j];
                if (vv > best) { best = vv; bidx = j0 + j; }
            }
            smax[tid] = best; sidx[tid] = bidx;
            __syncthreads();
            for (int s = NTHR / 2; s > 0; s >>= 1) {
                if (tid < s) {
                    const float v2 = smax[tid + s]; const int i2 = sidx[tid + s];
                    if (v2 > smax[tid] || (v2 == smax[tid] && i2 < sidx[tid])) {
                        smax[tid] = v2; sidx[tid] = i2;
                    }
                }
                __syncthreads();
            }
            if (tid == 0) s_tok = sidx[0];
            __syncthreads();
            const int tok = s_tok;
            const int j = tid * 4;
            *(u32x2*)(xb + b * HH + j) = *(const u32x2*)(Eb + (size_t)tok * HH + j);
        }
        grid_barrier(bar, phase);
    }
}

extern "C" void kernel_launch(void* const* d_in, const int* in_sizes, int n_in,
                              void* d_out, int out_size, void* d_ws, size_t ws_size,
                              hipStream_t stream) {
    const float* z     = (const float*)d_in[0];
    const float* E     = (const float*)d_in[1];
    const float* W_ih  = (const float*)d_in[2];
    const float* W_hh  = (const float*)d_in[3];
    const float* b_ih  = (const float*)d_in[4];
    const float* b_hh  = (const float*)d_in[5];
    const float* W_lat = (const float*)d_in[6];
    const float* b_lat = (const float*)d_in[7];
    const float* W_out = (const float*)d_in[8];
    const float* b_out = (const float*)d_in[9];
    float* out = (float*)d_out;

    char* ws = (char*)d_ws;
    size_t off = 0;
    auto carve = [&](size_t bytes) -> void* {
        void* p = ws + off;
        off = (off + bytes + 255) & ~(size_t)255;
        return p;
    };
    unsigned*       bar   = (unsigned*)      carve(256);
    unsigned short* zb    = (unsigned short*)carve((size_t)BB * LAT * 2);
    unsigned short* Eb    = (unsigned short*)carve((size_t)VV * HH * 2);
    unsigned short* Wihb  = (unsigned short*)carve((size_t)G4H * HH * 2);
    unsigned short* Whhb  = (unsigned short*)carve((size_t)G4H * HH * 2);
    unsigned short* Wlatb = (unsigned short*)carve((size_t)HH * LAT * 2);
    unsigned short* Woutb = (unsigned short*)carve((size_t)VV * HH * 2);
    float*          gates = (float*)         carve((size_t)BB * G4H * 4);
    float*          cbuf  = (float*)         carve((size_t)BB * HH * 4);
    unsigned short* hb    = (unsigned short*)carve((size_t)BB * HH * 2);
    unsigned short* xb    = (unsigned short*)carve((size_t)BB * HH * 2);
    (void)ws_size; (void)in_sizes; (void)n_in; (void)out_size;

    (void)hipMemsetAsync(bar, 0, 256, stream);

    // fp32 -> bf16 weight/activation conversion (once per launch; weights then live in L2)
    f32_to_bf16_kernel<<<512,  NTHR, 0, stream>>>(z,     zb,    BB * LAT);
    f32_to_bf16_kernel<<<4096, NTHR, 0, stream>>>(E,     Eb,    VV * HH);
    f32_to_bf16_kernel<<<4096, NTHR, 0, stream>>>(W_ih,  Wihb,  G4H * HH);
    f32_to_bf16_kernel<<<4096, NTHR, 0, stream>>>(W_hh,  Whhb,  G4H * HH);
    f32_to_bf16_kernel<<<1024, NTHR, 0, stream>>>(W_lat, Wlatb, HH * LAT);
    f32_to_bf16_kernel<<<4096, NTHR, 0, stream>>>(W_out, Woutb, VV * HH);

    // one persistent cooperative kernel runs the entire T=128 recurrence
    lstm_decoder_kernel<<<NWG, NTHR, 0, stream>>>(zb, Eb, Wihb, Whhb, Wlatb, Woutb,
                                                  b_ih, b_hh, b_lat, b_out,
                                                  gates, cbuf, hb, xb,
                                                  out, bar, TT);
}
